// KcatNet_27109833572443
// MI455X (gfx1250) — matrix operand order repaired and checked by reading, not compiled, not run on hardware
//
#include <hip/hip_runtime.h>

// ---------------------------------------------------------------------------
// KcatNet forward for MI455X (gfx1250, wave32, WMMA).
// All dense matmuls run through a strided/batched WMMA GEMM using
// v_wmma_f32_16x16x32_f16 (f32 -> f16 operands, f32 accumulate).
// Each wave computes a 32x32 C tile: 2 A-fragments x 2 B-fragments ->
// 4 WMMAs per 32-wide K step. Bounds handling is branchless (index clamp +
// value select) so all 64 loads per K step pipeline under one wait.
// Edge-message tensor [E,T,3FT] and RBF edge features are generated inside
// the GEMM A-loader (never materialized). Segment aggregation exploits
// dst = repeat(arange(N), 16) (16 contiguous edges per node, cnt==16).
// Workspace requirement: ~330 MB of float scratch in d_ws.
// Params assumed flattened in JAX pytree order (dict keys sorted).
// ---------------------------------------------------------------------------

typedef __attribute__((ext_vector_type(16))) _Float16 v16h;
typedef __attribute__((ext_vector_type(8)))  float    v8f;

#define BB   16
#define RR   512
#define AAT  48
#define HH   200
#define TT   5
#define FTT  40
#define NRES (BB*RR)     // 8192
#define NATM (BB*AAT)    // 768
#define DEGN 16
#define NEDG (NRES*DEGN) // 131072

// ---------------- A/B element loaders -------------------------------------
struct LdG {
  const float* p; long off, si, sj, sb;
  __device__ __forceinline__ float operator()(int i, int j, int z) const {
    return p[off + (long)z*sb + (long)i*si + (long)j*sj];
  }
  __device__ __forceinline__ void pf(int i, int j, int z) const {
    __builtin_prefetch((const void*)&p[off + (long)z*sb + (long)i*si + (long)j*sj], 0, 1);
  }
};

// PNA pre-MLP input: concat(x[dst], x[src], RBF(edge_weight)) per tower (z=t)
struct LdPna1 {
  const float* x; const int* srcI; const int* dstI; const float* ew;
  __device__ __forceinline__ float operator()(int e, int k, int t) const {
    // branchless: gather node feature (clamped k) and RBF, then select
    bool isx  = k < 2*FTT;
    int  node = (k < FTT) ? dstI[e] : srcI[e];
    int  kx   = (k < FTT) ? k : (k - FTT);
    kx = kx < FTT ? kx : 0;                       // clamp for RBF-range k
    float xv = x[(long)node*HH + t*FTT + kx];
    int j = t*FTT + (k - 2*FTT);
    float d = ew[e]; d = d < 1.f ? d : 1.f;
    float u = (d - (float)j*(1.0f/199.0f)) * 200.0f;   // sig = 1/200
    float rb = __expf(-u*u);
    return isx ? xv : rb;
  }
  __device__ __forceinline__ void pf(int, int, int) const {}
};

// PNA post input: concat(x_t, agg, agg*amp, agg*lin) per tower (z=t)
struct LdPnaPost {
  const float* x; const float* agg; const float* amp; const float* lin;
  __device__ __forceinline__ float operator()(int n, int k, int t) const {
    if (k < FTT) return x[(long)n*HH + t*FTT + k];
    int kk = k - FTT; int seg = kk / 160; int j = kk - seg*160;
    float v = agg[(long)n*800 + t*160 + j];
    float sc = (seg == 0) ? 1.f : (seg == 1 ? amp[n] : lin[n]);
    return v * sc;
  }
  __device__ __forceinline__ void pf(int, int, int) const {}
};

// ---------------- WMMA GEMM ------------------------------------------------
// One wave computes a 32x32 tile of C = A*B (+bias, relu, +add).
// VGPR layouts per CDNA5 ISA 7.12.2 (wave32, 16-bit A/B, f32 C/D):
//   A frag element e (0..15): K = koffA + (e&7) + ((e&8)<<1), koffA = half?8:0
//   B frag element e (0..15): K = koffB + (e&7) + ((e&8)<<1), koffB = half?8:0
//   C/D: col = lane&15, rows = (half?8:0)+r for VGPR r.
template <class LA>
__global__ __launch_bounds__(32)
void gemm_k(LA la, LdG lb, const float* bias, long biasSB,
            const float* add, long addSM, long addSB,
            float* C, long cOff, long cSM, long cSB,
            int M, int K, int N, int relu)
{
  const int lane = threadIdx.x & 31;
  const int half = lane >> 4;
  const int l15  = lane & 15;
  const int tm = blockIdx.y, tn = blockIdx.x, z = blockIdx.z;
  const int r0 = tm*32 + l15, r1 = r0 + 16;
  const int c0 = tn*32 + l15, c1 = c0 + 16;
  const int koffA = half ? 8 : 0;
  const int koffB = half ? 8 : 0;
  const int  rm0 = r0 < M ? r0 : M - 1;  const bool rv0 = r0 < M;
  const int  rm1 = r1 < M ? r1 : M - 1;  const bool rv1 = r1 < M;
  const int  cm0 = c0 < N ? c0 : N - 1;  const bool cv0 = c0 < N;
  const int  cm1 = c1 < N ? c1 : N - 1;  const bool cv1 = c1 < N;
  v8f a00 = {}, a01 = {}, a10 = {}, a11 = {};
  for (int k0 = 0; k0 < K; k0 += 32) {
    if (k0 + 32 < K) {           // prefetch next K step (global_prefetch_b8)
      la.pf(rm0, k0 + 32 + koffA, z);
      la.pf(rm1, k0 + 32 + koffA, z);
      lb.pf(k0 + 32 + koffB, cm0, z);
      lb.pf(k0 + 32 + koffB, cm1, z);
    }
    float fa0[16], fa1[16], fb0[16], fb1[16];
#pragma unroll
    for (int e = 0; e < 16; ++e) {
      int ka = k0 + koffA + (e & 7) + ((e & 8) << 1);
      int kb = k0 + koffB + (e & 7) + ((e & 8) << 1);
      bool kav = ka < K, kbv = kb < K;
      int kac = kav ? ka : 0;
      int kbc = kbv ? kb : 0;
      float va0 = la(rm0, kac, z);
      float va1 = la(rm1, kac, z);
      float vb0 = lb(kbc, cm0, z);
      float vb1 = lb(kbc, cm1, z);
      fa0[e] = (rv0 && kav) ? va0 : 0.f;
      fa1[e] = (rv1 && kav) ? va1 : 0.f;
      fb0[e] = (cv0 && kbv) ? vb0 : 0.f;
      fb1[e] = (cv1 && kbv) ? vb1 : 0.f;
    }
    v16h A0, A1, B0, B1;
#pragma unroll
    for (int e = 0; e < 16; ++e) {
      A0[e] = (_Float16)fa0[e];
      A1[e] = (_Float16)fa1[e];
      B0[e] = (_Float16)fb0[e];
      B1[e] = (_Float16)fb1[e];
    }
    a00 = __builtin_amdgcn_wmma_f32_16x16x32_f16(false, A0, false, B0, (short)0, a00, false, false);
    a01 = __builtin_amdgcn_wmma_f32_16x16x32_f16(false, A0, false, B1, (short)0, a01, false, false);
    a10 = __builtin_amdgcn_wmma_f32_16x16x32_f16(false, A1, false, B0, (short)0, a10, false, false);
    a11 = __builtin_amdgcn_wmma_f32_16x16x32_f16(false, A1, false, B1, (short)0, a11, false, false);
  }
  // epilogue: 4 sub-tiles
  const int rb = half ? 8 : 0;
#pragma unroll
  for (int cf = 0; cf < 2; ++cf) {
    const int  col = cf ? c1 : c0;
    const bool cv  = cf ? cv1 : cv0;
    if (!cv) continue;
    const float bvv = bias ? bias[(long)z*biasSB + col] : 0.f;
#pragma unroll
    for (int rf = 0; rf < 2; ++rf) {
      v8f acc = (rf == 0) ? (cf == 0 ? a00 : a01) : (cf == 0 ? a10 : a11);
      const int rowBase = tm*32 + rf*16 + rb;
#pragma unroll
      for (int r = 0; r < 8; ++r) {
        int row = rowBase + r;
        if (row < M) {
          float v = acc[r] + bvv;
          if (relu) v = v > 0.f ? v : 0.f;
          if (add)  v += add[(long)z*addSB + (long)row*addSM + col];
          C[cOff + (long)z*cSB + (long)row*cSM + col] = v;
        }
      }
    }
  }
}

template <class LA>
static inline void GEMM(hipStream_t st, LA la, LdG lb, const float* bias, long biasSB,
                        const float* add, long addSM, long addSB,
                        float* C, long cOff, long cSM, long cSB,
                        int M, int K, int N, int nb, int relu)
{
  gemm_k<LA><<<dim3((unsigned)((N+31)/32), (unsigned)((M+31)/32), (unsigned)nb),
               dim3(32), 0, st>>>(la, lb, bias, biasSB, add, addSM, addSB,
                                  C, cOff, cSM, cSB, M, K, N, relu);
}

// ---------------- elementwise / reduction kernels --------------------------
__global__ void k_deg_count(const int* dstI, float* cnt) {
  int e = blockIdx.x*blockDim.x + threadIdx.x;
  if (e < NEDG) atomicAdd(&cnt[dstI[e]], 1.0f);
}
__global__ void k_deg_finish(const float* cnt, float* dinv, float* amp, float* lin) {
  int i = blockIdx.x*blockDim.x + threadIdx.x;
  if (i >= NRES) return;
  float c = cnt[i];
  dinv[i] = rsqrtf(c + 1.0f);
  float cc = fmaxf(c, 1.0f);
  amp[i] = logf(cc + 1.0f) / logf((float)DEGN + 1.0f);
  lin[i] = cc / (float)DEGN;
}
__global__ void k_build_adj(const int* srcI, const int* dstI, float* adj) {
  int e = blockIdx.x*blockDim.x + threadIdx.x;
  if (e >= NEDG) return;
  int d = dstI[e], s = srcI[e];
  atomicAdd(&adj[(long)(d / RR)*RR*RR + (long)(d % RR)*RR + (s % RR)], 1.0f);
}
__global__ void k_dsum(const float* adj, float* dsum) {
  int i = blockIdx.x*blockDim.x + threadIdx.x;
  if (i >= BB*RR) return;
  const float* row = adj + (long)i*RR;
  float s = 0.f;
  for (int q = 0; q < RR; ++q) s += row[q];
  dsum[i] = s;
}
__global__ void k_embed_add(float* xa, const int* ids, const float* emb) {
  int idx = blockIdx.x*blockDim.x + threadIdx.x;
  if (idx >= NATM*HH) return;
  int a = idx / HH, h = idx % HH;
  xa[idx] += emb[(long)ids[a]*HH + h];
}
__global__ void k_graphnorm(float* x, const float* a, const float* g, const float* b,
                            int nper) {
  int bi = blockIdx.y;
  int h  = blockIdx.x*blockDim.x + threadIdx.x;
  if (h >= HH) return;
  const long base = (long)bi*nper*HH + h;
  float m = 0.f;
  for (int i = 0; i < nper; ++i) m += x[base + (long)i*HH];
  m /= nper;
  float am = a[h]*m, v = 0.f;
  for (int i = 0; i < nper; ++i) { float s = x[base + (long)i*HH] - am; v += s*s; }
  float inv = rsqrtf(v / nper + 1e-5f);
  for (int i = 0; i < nper; ++i) {
    float s = x[base + (long)i*HH] - am;
    x[base + (long)i*HH] = g[h]*s*inv + b[h];
  }
}
__global__ void k_layernorm(const float* in, long ldin, float* out, long ldout,
                            long coloff, const float* g, const float* b, int rows) {
  int r = blockIdx.x; if (r >= rows) return;
  int t = threadIdx.x;
  float s = 0.f;
  for (int h = t; h < HH; h += 32) s += in[(long)r*ldin + h];
  for (int o = 16; o > 0; o >>= 1) s += __shfl_down(s, o);
  float m = __shfl(s, 0) / HH;
  float v = 0.f;
  for (int h = t; h < HH; h += 32) { float d = in[(long)r*ldin + h] - m; v += d*d; }
  for (int o = 16; o > 0; o >>= 1) v += __shfl_down(v, o);
  float inv = rsqrtf(__shfl(v, 0) / HH + 1e-5f);
  for (int h = t; h < HH; h += 32)
    out[(long)r*ldout + coloff + h] = g[h]*(in[(long)r*ldin + h] - m)*inv + b[h];
}
__global__ void k_pna_agg(const float* h2, const float* cnt, float* agg) {
  int idx = blockIdx.x*blockDim.x + threadIdx.x;
  if (idx >= NRES*HH) return;
  int n = idx / HH, c = idx % HH;
  int t = c / FTT, f = c % FTT;
  float s1 = 0.f, s2 = 0.f, mn = 3.4e38f, mx = -3.4e38f;
  long base = (long)n*DEGN*HH + c;
  for (int j = 0; j < DEGN; ++j) {
    float v = h2[base + (long)j*HH];
    s1 += v; s2 += v*v; mn = fminf(mn, v); mx = fmaxf(mx, v);
  }
  float c1 = fmaxf(cnt[n], 1.f);
  float mean = s1 / c1;
  float stdv = sqrtf(fmaxf(s2 / c1 - mean*mean, 0.f) + 1e-5f);
  long o = (long)n*800 + t*160 + f;
  agg[o] = mean; agg[o + 40] = mn; agg[o + 80] = mx; agg[o + 120] = stdv;
}
__global__ void k_gcn_agg(const float* hw, const int* srcI, const float* dinv,
                          const float* bias, float* out, int C, int relu) {
  int idx = blockIdx.x*blockDim.x + threadIdx.x;
  if (idx >= NRES*C) return;
  int n = idx / C, c = idx % C;
  float dn = dinv[n];
  float s = hw[(long)n*C + c]*dn*dn;
  for (int j = 0; j < DEGN; ++j) {
    int si = srcI[n*DEGN + j];
    s += hw[(long)si*C + c]*dinv[si]*dn;
  }
  s += bias[c];
  out[(long)n*C + c] = relu ? fmaxf(s, 0.f) : s;
}
__global__ void k_softmax(float* x, int ngroups, int cnt, int gi,
                          long sOuter, long sInner, long stride) {
  int g = blockIdx.x*blockDim.x + threadIdx.x;
  if (g >= ngroups) return;
  long base = (long)(g / gi)*sOuter + (long)(g % gi)*sInner;
  float mx = -3.4e38f;
  for (int i = 0; i < cnt; ++i) mx = fmaxf(mx, x[base + (long)i*stride]);
  float s = 0.f;
  for (int i = 0; i < cnt; ++i) {
    float e = __expf(x[base + (long)i*stride] - mx);
    x[base + (long)i*stride] = e; s += e;
  }
  float inv = 1.0f / s;
  for (int i = 0; i < cnt; ++i) x[base + (long)i*stride] *= inv;
}
__global__ void k_motif_pool(const float* attn, const float* xa, float* molx) {
  int idx = blockIdx.x*blockDim.x + threadIdx.x;
  if (idx >= BB*HH) return;
  int b = idx / HH, h = idx % HH, t = (h / FTT);
  float s = 0.f;
  for (int a = 0; a < AAT; ++a)
    s += attn[(long)b*AAT*TT + a*TT + t] * xa[((long)b*AAT + a)*HH + h];
  molx[idx] = s;
}
__global__ void k_concat2(const float* a, const float* b2, float* out, int rows, int W) {
  int idx = blockIdx.x*blockDim.x + threadIdx.x;
  if (idx >= rows*2*W) return;
  int r = idx / (2*W), c = idx % (2*W);
  out[idx] = (c < W) ? a[(long)r*W + c] : b2[(long)r*W + (c - W)];
}
__global__ void k_alpha(const float* q, const float* kk, float* alpha, int K) {
  int idx = blockIdx.x*blockDim.x + threadIdx.x;
  if (idx >= BB*K*TT) return;
  int b = idx / (K*TT); int rem = idx % (K*TT); int k = rem / TT, t = rem % TT;
  float s = 0.f;
  for (int f = 0; f < FTT; ++f)
    s += q[(long)b*HH + t*FTT + f] * kk[((long)b*K + k)*HH + t*FTT + f];
  alpha[idx] = s * 0.15811388300841897f;   // 1/sqrt(FT)
}
__global__ void k_aggv(const float* alpha, const float* vv, float* out, int K) {
  int idx = blockIdx.x*blockDim.x + threadIdx.x;
  if (idx >= BB*HH) return;
  int b = idx / HH, h = idx % HH, t = h / FTT;
  float s = 0.f;
  for (int k = 0; k < K; ++k)
    s += alpha[(long)b*K*TT + k*TT + t] * vv[((long)b*K + k)*HH + h];
  out[idx] = s;
}
__global__ void k_cluster_badd(float* cx, const float* t16, int K) {
  int idx = blockIdx.x*blockDim.x + threadIdx.x;
  if (idx >= BB*K*HH) return;
  int b = idx / (K*HH), h = idx % HH;
  cx[idx] += t16[(long)b*HH + h];
}
__global__ void k_atom_badd(float* xa, const float* t16) {
  int idx = blockIdx.x*blockDim.x + threadIdx.x;
  if (idx >= NATM*HH) return;
  int b = idx / (AAT*HH), h = idx % HH;
  xa[idx] += t16[(long)b*HH + h];
}
__global__ void k_copycol(const float* in, int W, float* out, long ldo, long coloff,
                          int rows) {
  int idx = blockIdx.x*blockDim.x + threadIdx.x;
  if (idx >= rows*W) return;
  int r = idx / W, c = idx % W;
  out[(long)r*ldo + coloff + c] = in[idx];
}
__global__ void k_wrowmax(const float* x, const float* w, float* out,
                          int nper, long ldx, long ldo, long coloff) {
  int idx = blockIdx.x*blockDim.x + threadIdx.x;
  if (idx >= BB*HH) return;
  int b = idx / HH, h = idx % HH;
  float m = -3.4e38f;
  for (int i = 0; i < nper; ++i) {
    float v = x[((long)b*nper + i)*ldx + h];
    if (w) v *= w[(long)b*nper + i];
    m = fmaxf(m, v);
  }
  out[(long)b*ldo + coloff + h] = m;
}
__global__ void k_talpha(const float* alpha, const float* wvec, float* out, int K) {
  int idx = blockIdx.x*blockDim.x + threadIdx.x;
  if (idx >= BB*K) return;
  float s = 0.f;
  for (int t = 0; t < TT; ++t) s += alpha[(long)idx*TT + t]*fabsf(wvec[t]);
  out[idx] = s;
}
__global__ void k_rscval(const float* s, const float* ta, float* val, int K) {
  int idx = blockIdx.x*blockDim.x + threadIdx.x;
  if (idx >= BB*RR) return;
  int b = idx / RR;
  float v = 0.f;
  for (int k = 0; k < K; ++k) v += s[(long)idx*K + k]*ta[(long)b*K + k];
  val[idx] = v;
}
__device__ __forceinline__ float blockReduce64(float v, float* sh) {
  int t = threadIdx.x;
  sh[t] = v; __syncthreads();
  for (int o = 32; o > 0; o >>= 1) { if (t < o) sh[t] += sh[t + o]; __syncthreads(); }
  float r = sh[0]; __syncthreads();
  return r;
}
__global__ void k_mincut(const float* s, const float* oadj, const float* ss,
                         const float* dsum, float* loss, int K) {
  __shared__ float sh[64];
  int b = blockIdx.x, t = threadIdx.x;
  float num = 0.f;
  for (int k = t; k < K; k += 64) num += oadj[(long)b*K*K + (long)k*K + k];
  num = blockReduce64(num, sh);
  float den = 0.f;
  for (int r = t; r < RR; r += 64) {
    const float* sr = s + ((long)b*RR + r)*K;
    float q = 0.f;
    for (int k = 0; k < K; ++k) q += sr[k]*sr[k];
    den += dsum[(long)b*RR + r]*q;
  }
  den = blockReduce64(den, sh);
  float f2 = 0.f;
  for (int i = t; i < K*K; i += 64) { float v = ss[(long)b*K*K + i]; f2 += v*v; }
  f2 = blockReduce64(f2, sh);
  float frob = sqrtf(f2) + 1e-9f;
  float diag = rsqrtf((float)K);
  float o2 = 0.f;
  for (int i = t; i < K*K; i += 64) {
    float v = ss[(long)b*K*K + i]/frob - ((i / K == i % K) ? diag : 0.f);
    o2 += v*v;
  }
  o2 = blockReduce64(o2, sh);
  if (t == 0)
    atomicAdd(loss, (-(num / (den + 1e-9f)) + sqrtf(o2)) * (1.0f/BB));
}

// ---------------------------------------------------------------------------
extern "C" void kernel_launch(void* const* d_in, const int* in_sizes, int n_in,
                              void* d_out, int out_size, void* d_ws, size_t ws_size,
                              hipStream_t stream) {
  (void)in_sizes; (void)n_in; (void)out_size; (void)ws_size;
  auto F = [&](int i){ return (const float*)d_in[i]; };
  auto I = [&](int i){ return (const int*)d_in[i]; };
  auto g1 = [](long n){ return dim3((unsigned)((n + 255)/256)); };

  const int*   mol_x      = I(0);
  const float* mol_x_feat = F(1);
  const float* mol_total  = F(2);
  const float* esm        = F(3);
  const float* prot5      = F(4);
  const int*   srcI       = I(5);
  const int*   dstI       = I(5) + NEDG;
  const float* ew         = F(6);
  // params: JAX pytree flatten (keys sorted alphabetically, recursively)
  const float* af_b = F(9);  const float* af_w = F(10);
  const float* emb  = F(11);
  const float* attn1 = F(12); const float* attn2 = F(13);
  const float* c0_b = F(14); const float* c0_w = F(15);
  const float* c1_b = F(16); const float* c1_w = F(17);
  const float* c2_b = F(18); const float* c2_w = F(19);
  const float* cfu_b = F(20); const float* cfu_w = F(21);
  const float* esm_b = F(22); const float* esm_w = F(23);
  const float* gn_a = F(24); const float* gn_b = F(25); const float* gn_g = F(26);
  const int LBASE = 27, LSZ = 31;          // layers occupy 27..119
  const float* ln_b = F(120); const float* ln_g = F(121);
  const float* me2_b = F(122); const float* me2_w = F(123);
  const float* me_b = F(124); const float* me_w = F(125);
  const float* mfu_b = F(126); const float* mfu_w = F(127);
  const float* p5_b = F(128); const float* p5_w = F(129);
  const float* rfu2_b = F(130); const float* rfu2_w = F(131);
  const float* rfu_b = F(132); const float* rfu_w = F(133);
  const float* seq_b = F(134); const float* seq_w = F(135);

  // ---- workspace carve (floats) ----
  float* Wf = (float*)d_ws;
  size_t off = 0;
  auto alloc = [&](size_t n){ float* p = Wf + off; off += (n + 63) & ~(size_t)63; return p; };
  float* f_rini = alloc((size_t)NRES*800);
  float* f_x    = alloc((size_t)NRES*HH);
  float* f_adj  = alloc((size_t)BB*RR*RR);
  float* f_deg  = alloc(NRES);
  float* f_dinv = alloc(NRES);
  float* f_amp  = alloc(NRES);
  float* f_lin  = alloc(NRES);
  float* f_dsum = alloc(BB*RR);
  float* f_h1   = alloc((size_t)NEDG*HH);
  float* f_h2   = alloc((size_t)NEDG*HH);
  float* f_agg  = alloc((size_t)NRES*800);
  float* f_po   = alloc((size_t)NRES*HH);
  float* f_hw   = alloc((size_t)NRES*400);
  float* f_hg   = alloc((size_t)NRES*400);
  float* f_s    = alloc((size_t)NRES*20);
  float* f_As   = alloc((size_t)BB*RR*20);
  float* f_oadj = alloc(BB*20*20);
  float* f_ss   = alloc(BB*20*20);
  float* f_cx   = alloc((size_t)BB*20*HH);
  float* f_sc   = alloc((size_t)NRES*HH);
  float* f_xa   = alloc((size_t)NATM*HH);
  float* f_xa2  = alloc((size_t)NATM*HH);
  float* f_mt   = alloc(BB*HH);
  float* f_attn = alloc(NATM*TT);
  float* f_q    = alloc(BB*HH);
  float* f_kk   = alloc((size_t)BB*20*HH);
  float* f_vv   = alloc((size_t)BB*20*HH);
  float* f_alpha= alloc(BB*20*TT);
  float* f_molx = alloc(BB*HH);
  float* f_ta   = alloc(BB*20);
  float* f_rsc  = alloc(BB*RR);
  float* f_rfc  = alloc(BB*600);
  float* f_rf2c = alloc(BB*600);
  float* f_cfc  = alloc(BB*600);
  float* f_mfc  = alloc(BB*600);
  float* f_feat = alloc(BB*800);
  float* f_fh1  = alloc(BB*512);
  float* f_fh2  = alloc(BB*128);
  float* t0 = alloc(BB*HH); float* t1 = alloc(BB*HH); float* t2 = alloc(BB*HH);
  float* t3 = alloc(BB*HH); float* t4 = alloc(BB*HH); float* t5 = alloc(BB*HH);
  float* f_cat2 = alloc(BB*400);

  hipMemsetAsync(d_out, 0, 17*sizeof(float), stream);
  hipMemsetAsync(f_deg, 0, (size_t)NRES*sizeof(float), stream);
  hipMemsetAsync(f_adj, 0, (size_t)BB*RR*RR*sizeof(float), stream);

  // ---- graph setup ----
  k_deg_count<<<g1(NEDG),256,0,stream>>>(dstI, f_deg);
  k_build_adj<<<g1(NEDG),256,0,stream>>>(srcI, dstI, f_adj);
  k_deg_finish<<<g1(NRES),256,0,stream>>>(f_deg, f_dinv, f_amp, f_lin);
  k_dsum<<<g1(BB*RR),256,0,stream>>>(f_adj, f_dsum);

  // ---- sequence embeddings ----
  GEMM(stream, LdG{prot5,0,1280,1,0}, LdG{p5_w,0,400,1,0}, p5_b,0, nullptr,0,0,
       f_rini, 0, 800, 0, NRES, 1280, 400, 1, 1);
  GEMM(stream, LdG{esm,0,40,1,0}, LdG{esm_w,0,400,1,0}, esm_b,0, nullptr,0,0,
       f_rini, 400, 800, 0, NRES, 40, 400, 1, 1);
  GEMM(stream, LdG{f_rini,0,800,1,0}, LdG{seq_w,0,200,1,0}, seq_b,0, nullptr,0,0,
       f_x, 0, 200, 0, NRES, 800, 200, 1, 1);

  // ---- atom init ----
  GEMM(stream, LdG{mol_x_feat,0,43,1,0}, LdG{af_w,0,200,1,0}, af_b,0, nullptr,0,0,
       f_xa, 0, 200, 0, NATM, 43, 200, 1, 1);
  k_embed_add<<<g1((long)NATM*HH),256,0,stream>>>(f_xa, mol_x, emb);

  // ---- molecule total features ----
  GEMM(stream, LdG{mol_total,0,1024,1,0}, LdG{me_w,0,200,1,0}, me_b,0, nullptr,0,0,
       t0, 0, 200, 0, BB, 1024, 200, 1, 1);
  GEMM(stream, LdG{t0,0,200,1,0}, LdG{me2_w,0,200,1,0}, me2_b,0, nullptr,0,0,
       t1, 0, 200, 0, BB, 200, 200, 1, 0);
  k_layernorm<<<BB,32,0,stream>>>(t1, 200, f_mt, 200, 0, ln_g, ln_b, BB);

  float* xa_cur = f_xa; float* xa_nxt = f_xa2;
  const int KSarr[3] = {10, 15, 20};
  for (int idx = 0; idx < 3; ++idx) {
    const int K = KSarr[idx];
    const int LB = LBASE + LSZ*idx;
    const float* aet2_b = F(LB+0);  const float* aet2_w = F(LB+1);
    const float* aet_b  = F(LB+2);  const float* aet_w  = F(LB+3);
    const float* au_b   = F(LB+4);  const float* au_w   = F(LB+5);
    const float* clu_b1 = F(LB+6);  const float* clu_b2 = F(LB+7);
    const float* clu_w1 = F(LB+8);  const float* clu_w2 = F(LB+9);
    const float* gate_b = F(LB+10); const float* gate_w = F(LB+11);
    const float* i_wc   = F(LB+12); const float* i_wcb  = F(LB+13);
    const float* i_wk   = F(LB+14); const float* i_wo   = F(LB+15);
    const float* i_wob  = F(LB+16); const float* i_wq   = F(LB+17);
    const float* i_wv   = F(LB+18);
    const float* mu_b   = F(LB+19); const float* mu_w   = F(LB+20);
    const float* mix_b  = F(LB+21); const float* mix_w  = F(LB+22);
    const float* post_b = F(LB+23); const float* post_w = F(LB+24);
    const float* pre_b1 = F(LB+25); const float* pre_b2 = F(LB+26);
    const float* pre_w1 = F(LB+27); const float* pre_w2 = F(LB+28);
    const float* ru_b   = F(LB+29); const float* ru_w   = F(LB+30);

    // GraphNorm
    k_graphnorm<<<dim3((HH+63)/64, BB),64,0,stream>>>(f_x, gn_a, gn_g, gn_b, RR);
    k_graphnorm<<<dim3((HH+63)/64, BB),64,0,stream>>>(xa_cur, gn_a, gn_g, gn_b, AAT);

    // PNA (towers batched over z)
    GEMM(stream, LdPna1{f_x, srcI, dstI, ew}, LdG{pre_w1,0,40,1,4800}, pre_b1,40,
         nullptr,0,0, f_h1, 0, 200, 40, NEDG, 120, 40, TT, 1);
    GEMM(stream, LdG{f_h1,0,200,1,40}, LdG{pre_w2,0,40,1,1600}, pre_b2,40,
         nullptr,0,0, f_h2, 0, 200, 40, NEDG, 40, 40, TT, 0);
    k_pna_agg<<<g1((long)NRES*HH),256,0,stream>>>(f_h2, f_deg, f_agg);
    GEMM(stream, LdPnaPost{f_x, f_agg, f_amp, f_lin}, LdG{post_w,0,40,1,20800},
         post_b,40, nullptr,0,0, f_po, 0, 200, 40, NRES, 520, 40, TT, 0);
    GEMM(stream, LdG{f_po,0,200,1,0}, LdG{mix_w,0,200,1,0}, mix_b,0, nullptr,0,0,
         f_x, 0, 200, 0, NRES, 200, 200, 1, 0);
    k_wrowmax<<<g1(BB*HH),256,0,stream>>>(f_x, nullptr, f_rfc, RR, 200, 600, (long)idx*200);

    // GCN cluster assignment
    GEMM(stream, LdG{f_x,0,200,1,0}, LdG{clu_w1,0,400,1,0}, nullptr,0, nullptr,0,0,
         f_hw, 0, 400, 0, NRES, 200, 400, 1, 0);
    k_gcn_agg<<<g1((long)NRES*400),256,0,stream>>>(f_hw, srcI, f_dinv, clu_b1, f_hg, 400, 1);
    GEMM(stream, LdG{f_hg,0,400,1,0}, LdG{clu_w2,0,K,1,0}, nullptr,0, nullptr,0,0,
         f_hw, 0, K, 0, NRES, 400, K, 1, 0);
    k_gcn_agg<<<g1((long)NRES*K),256,0,stream>>>(f_hw, srcI, f_dinv, clu_b2, f_s, K, 0);
    k_softmax<<<g1(NRES),256,0,stream>>>(f_s, NRES, K, 1, K, 0, 1);

    // mincut pooling (batched einsums via strided GEMM)
    GEMM(stream, LdG{f_s,0,1,K,(long)RR*K}, LdG{f_x,0,200,1,(long)RR*200}, nullptr,0,
         nullptr,0,0, f_cx, 0, 200, (long)K*200, K, RR, HH, BB, 0);
    GEMM(stream, LdG{f_adj,0,RR,1,(long)RR*RR}, LdG{f_s,0,K,1,(long)RR*K}, nullptr,0,
         nullptr,0,0, f_As, 0, K, (long)RR*K, RR, RR, K, BB, 0);
    GEMM(stream, LdG{f_s,0,1,K,(long)RR*K}, LdG{f_As,0,K,1,(long)RR*K}, nullptr,0,
         nullptr,0,0, f_oadj, 0, K, (long)K*K, K, RR, K, BB, 0);
    GEMM(stream, LdG{f_s,0,1,K,(long)RR*K}, LdG{f_s,0,K,1,(long)RR*K}, nullptr,0,
         nullptr,0,0, f_ss, 0, K, (long)K*K, K, RR, K, BB, 0);
    k_mincut<<<BB,64,0,stream>>>(f_s, f_oadj, f_ss, f_dsum, ((float*)d_out)+16, K);
    k_layernorm<<<BB*K,32,0,stream>>>(f_cx, 200, f_cx, 200, 0, ln_g, ln_b, BB*K);

    // atom update + MotifPool
    GEMM(stream, LdG{xa_cur,0,200,1,0}, LdG{au_w,0,200,1,0}, au_b,0, nullptr,0,0,
         xa_nxt, 0, 200, 0, NATM, 200, 200, 1, 1);
    GEMM(stream, LdG{xa_nxt,0,200,1,0}, LdG{gate_w,0,TT,1,0}, gate_b,0, nullptr,0,0,
         f_attn, 0, TT, 0, NATM, 200, TT, 1, 0);
    k_softmax<<<g1(BB*TT),256,0,stream>>>(f_attn, BB*TT, AAT, TT, (long)AAT*TT, 1, TT);
    k_motif_pool<<<g1(BB*HH),256,0,stream>>>(f_attn, xa_nxt, t0);
    k_layernorm<<<BB,32,0,stream>>>(t0, 200, t1, 200, 0, ln_g, ln_b, BB);
    k_concat2<<<g1(BB*400),256,0,stream>>>(f_mt, t1, f_cat2, BB, HH);
    GEMM(stream, LdG{f_cat2,0,400,1,0}, LdG{aet_w,0,200,1,0}, aet_b,0, nullptr,0,0,
         t2, 0, 200, 0, BB, 400, 200, 1, 1);
    GEMM(stream, LdG{t2,0,200,1,0}, LdG{aet2_w,0,200,1,0}, aet2_b,0, nullptr,0,0,
         t3, 0, 200, 0, BB, 200, 200, 1, 0);
    k_layernorm<<<BB,32,0,stream>>>(t3, 200, f_molx, 200, 0, ln_g, ln_b, BB);

    // InterConv (clusters -> molecule cross attention)
    GEMM(stream, LdG{f_molx,0,200,1,0}, LdG{i_wq,0,200,1,0}, nullptr,0, nullptr,0,0,
         f_q, 0, 200, 0, BB, 200, 200, 1, 0);
    GEMM(stream, LdG{f_cx,0,200,1,0}, LdG{i_wk,0,200,1,0}, nullptr,0, nullptr,0,0,
         f_kk, 0, 200, 0, BB*K, 200, 200, 1, 0);
    GEMM(stream, LdG{f_cx,0,200,1,0}, LdG{i_wv,0,200,1,0}, nullptr,0, nullptr,0,0,
         f_vv, 0, 200, 0, BB*K, 200, 200, 1, 0);
    k_alpha<<<g1((long)BB*K*TT),256,0,stream>>>(f_q, f_kk, f_alpha, K);
    k_softmax<<<g1(BB*TT),256,0,stream>>>(f_alpha, BB*TT, K, TT, (long)K*TT, 1, TT);
    k_aggv<<<g1(BB*HH),256,0,stream>>>(f_alpha, f_vv, t4, K);
    GEMM(stream, LdG{t4,0,200,1,0}, LdG{i_wo,0,200,1,0}, i_wob,0, f_molx,200,0,
         f_molx, 0, 200, 0, BB, 200, 200, 1, 0);
    GEMM(stream, LdG{f_molx,0,200,1,0}, LdG{i_wc,0,200,1,0}, i_wcb,0, nullptr,0,0,
         t5, 0, 200, 0, BB, 200, 200, 1, 1);
    k_cluster_badd<<<g1((long)BB*K*HH),256,0,stream>>>(f_cx, t5, K);
    k_copycol<<<g1(BB*HH),256,0,stream>>>(f_molx, HH, f_mfc, 600, (long)idx*200, BB);

    // atom broadcast update
    GEMM(stream, LdG{f_molx,0,200,1,0}, LdG{mu_w,0,200,1,0}, mu_b,0, nullptr,0,0,
         t0, 0, 200, 0, BB, 200, 200, 1, 1);
    k_atom_badd<<<g1((long)NATM*HH),256,0,stream>>>(xa_nxt, t0);

    // cluster feature readout (csc weighted max + LN)
    k_talpha<<<g1(BB*K),256,0,stream>>>(f_alpha, attn1, f_ta, K);
    k_softmax<<<g1(BB),256,0,stream>>>(f_ta, BB, K, 1, K, 0, 1);
    k_wrowmax<<<g1(BB*HH),256,0,stream>>>(f_cx, f_ta, t0, K, 200, 200, 0);
    k_layernorm<<<BB,32,0,stream>>>(t0, 200, f_cfc, 600, (long)idx*200, ln_g, ln_b, BB);

    // residue update: x += relu((s @ cluster_x) @ ru_w + b)
    GEMM(stream, LdG{f_s,0,K,1,(long)RR*K}, LdG{f_cx,0,200,1,(long)K*200}, nullptr,0,
         nullptr,0,0, f_sc, 0, 200, (long)RR*200, RR, K, HH, BB, 0);
    GEMM(stream, LdG{f_sc,0,200,1,0}, LdG{ru_w,0,200,1,0}, ru_b,0, f_x,200,0,
         f_x, 0, 200, 0, NRES, 200, 200, 1, 1);

    // residue feature readout (rsc weighted max + LN)
    k_talpha<<<g1(BB*K),256,0,stream>>>(f_alpha, attn2, f_ta, K);
    k_rscval<<<g1(BB*RR),256,0,stream>>>(f_s, f_ta, f_rsc, K);
    k_softmax<<<g1(BB),256,0,stream>>>(f_rsc, BB, RR, 1, RR, 0, 1);
    k_wrowmax<<<g1(BB*HH),256,0,stream>>>(f_x, f_rsc, t0, RR, 200, 200, 0);
    k_layernorm<<<BB,32,0,stream>>>(t0, 200, f_rf2c, 600, (long)idx*200, ln_g, ln_b, BB);

    float* tmp = xa_cur; xa_cur = xa_nxt; xa_nxt = tmp;
  }

  // ---- fusion + classifier head ----
  GEMM(stream, LdG{f_mfc,0,600,1,0}, LdG{mfu_w,0,200,1,0}, mfu_b,0, nullptr,0,0,
       t0, 0, 200, 0, BB, 600, 200, 1, 1);
  GEMM(stream, LdG{f_rfc,0,600,1,0}, LdG{rfu_w,0,200,1,0}, rfu_b,0, nullptr,0,0,
       t1, 0, 200, 0, BB, 600, 200, 1, 1);
  GEMM(stream, LdG{f_rf2c,0,600,1,0}, LdG{rfu2_w,0,200,1,0}, rfu2_b,0, nullptr,0,0,
       t2, 0, 200, 0, BB, 600, 200, 1, 1);
  GEMM(stream, LdG{f_cfc,0,600,1,0}, LdG{cfu_w,0,200,1,0}, cfu_b,0, nullptr,0,0,
       t3, 0, 200, 0, BB, 600, 200, 1, 1);
  k_copycol<<<g1(BB*HH),256,0,stream>>>(t1, HH, f_feat, 800, 0,   BB);
  k_copycol<<<g1(BB*HH),256,0,stream>>>(t2, HH, f_feat, 800, 200, BB);
  k_copycol<<<g1(BB*HH),256,0,stream>>>(t3, HH, f_feat, 800, 400, BB);
  k_copycol<<<g1(BB*HH),256,0,stream>>>(t0, HH, f_feat, 800, 600, BB);
  GEMM(stream, LdG{f_feat,0,800,1,0}, LdG{c0_w,0,512,1,0}, c0_b,0, nullptr,0,0,
       f_fh1, 0, 512, 0, BB, 800, 512, 1, 1);
  GEMM(stream, LdG{f_fh1,0,512,1,0}, LdG{c1_w,0,128,1,0}, c1_b,0, nullptr,0,0,
       f_fh2, 0, 128, 0, BB, 512, 128, 1, 1);
  GEMM(stream, LdG{f_fh2,0,128,1,0}, LdG{c2_w,0,1,1,0}, c2_b,0, nullptr,0,0,
       (float*)d_out, 0, 1, 0, BB, 128, 1, 1, 0);
}
